// VanillaMAPPOMicroCommResidualAdapterAgent_60619168416476
// MI455X (gfx1250) — matrix operand order repaired and checked by reading, not compiled, not run on hardware
//
#include <hip/hip_runtime.h>
#include <hip/hip_bf16.h>

typedef __bf16 bf16_t;
typedef __attribute__((ext_vector_type(16))) __bf16 v16bf;
typedef __attribute__((ext_vector_type(8)))  float  v8f;

#define ROWS  131072     // BS*N
#define BS_   4096
#define NA    32
#define DIN   128
#define HDIM  256
#define HEADS 4
#define HD    16
#define VD    8
#define KTOP  8
#define FC    32
#define ACTD  20
#define GDIM  288        // H + FC

__device__ __forceinline__ float bf2f(bf16_t x) {
    unsigned short us = __builtin_bit_cast(unsigned short, x);
    unsigned u = ((unsigned)us) << 16;
    return __builtin_bit_cast(float, u);
}
__device__ __forceinline__ bf16_t f2bf(float f) {
    unsigned u = __builtin_bit_cast(unsigned, f);
    unsigned r = u + 0x7FFFu + ((u >> 16) & 1u);   // round-to-nearest-even
    return __builtin_bit_cast(bf16_t, (unsigned short)(r >> 16));
}

// Packed-A layout (ISA 16-bit A 16x32 fragment order, contiguous 32B per lane):
//   idx(row, k; Kd) = ((strip*(Kd/32) + kstep)*32 + lane)*16 + e
//   strip=row>>4, kstep=k>>5, kk=k&31, half=(kk>>3)&1, e=(kk&7)+8*(kk>>4),
//   lane = half*16 + (row&15)
__device__ __forceinline__ long packA_idx(int row, int c, int KdTot) {
    const int kk   = c & 31;
    const int e    = (kk & 7) + ((kk >> 4) << 3);
    const int lane = (((kk >> 3) & 1) << 4) + (row & 15);
    return ((((long)(row >> 4) * (KdTot >> 5)) + (c >> 5)) * 32 + lane) * 16 + e;
}

// ---------------------------------------------------------------- activation f32 -> packed-A bf16
__global__ void cvt_packA(const float* __restrict__ s, bf16_t* __restrict__ d, int M, int Kd) {
    long i = (long)blockIdx.x * 256 + threadIdx.x;
    if (i >= (long)M * Kd) return;
    int r = (int)(i / Kd), c = (int)(i % Kd);
    d[packA_idx(r, c, Kd)] = f2bf(s[i]);
}

// ---------------------------------------------------------------- weight f32 -> packed-B bf16
// B fragment order: idx = ((kstep*(Nout/16) + nt)*32 + lane)*16 + e
//   k = kstep*32 + (lane>>4)*8 + (e&7) + (e>>3)*16 ; col = nt*16 + (lane&15)
// srcCols < Nout zero-pads extra columns (used for Wp2 20 -> 32).
__global__ void pack_w(const float* __restrict__ W, bf16_t* __restrict__ out,
                       int Kd, int Nout, int srcCols) {
    long o = (long)blockIdx.x * 256 + threadIdx.x;
    if (o >= (long)Kd * Nout) return;
    const int nt16 = Nout >> 4;
    int e    = (int)(o & 15);
    int lane = (int)((o >> 4) & 31);
    long t   = o >> 9;
    int nt   = (int)(t % nt16);
    int ks   = (int)(t / nt16);
    int k    = ks * 32 + ((lane >> 4) << 3) + (e & 7) + ((e >> 3) << 4);
    int col  = (nt << 4) + (lane & 15);
    out[o] = (col < srcCols) ? f2bf(W[(long)k * srcCols + col]) : f2bf(0.f);
}

// ---------------------------------------------------------------- generic bf16 WMMA GEMM
// C[M x Nout] = act(A @ W + bias); A,B pre-swizzled into WMMA fragment layouts.
// Optional epilogue: val = resid[row*ldc+col] + val * (scale * gatev[row]).
// Outputs: outF (f32 row-major, ldc/ncols), outP (bf16 packed-A with Kd=Nout),
//          outR (bf16 row-major, width Nout).
__global__ void gemm_bf16_wmma(
    const bf16_t* __restrict__ Ap, const bf16_t* __restrict__ Bp,
    const float* __restrict__ bias,
    int M, int Kd, int Nout,
    float* __restrict__ outF, int ldc, int ncols,
    bf16_t* __restrict__ outP, bf16_t* __restrict__ outR,
    int act,
    const float* __restrict__ resid, const float* __restrict__ gatev, float scale)
{
    const int lane   = threadIdx.x & 31;
    const int wave   = threadIdx.x >> 5;
    const int ntiles = Nout >> 5;                       // 32-wide wave tiles
    const long tile  = (long)blockIdx.x * 8 + wave;
    const long total = (long)(M >> 4) * ntiles;
    if (tile >= total) return;                          // uniform per wave
    const int strip = (int)(tile / ntiles);
    const int ntile = (int)(tile % ntiles);
    const int ksteps = Kd >> 5;
    const int nt16   = Nout >> 4;

    long abase = (((long)strip * ksteps) * 32 + lane) * 16;
    long bbase = (((long)(ntile * 2)) * 32 + lane) * 16;
    const long bstep = (long)nt16 * 512;                // per-kstep B advance

    v8f acc0 = {}; v8f acc1 = {};
    for (int ks = 0; ks < ksteps; ++ks) {
        v16bf a  = *(const v16bf*)(const void*)(Ap + abase);
        v16bf b0 = *(const v16bf*)(const void*)(Bp + bbase);
        v16bf b1 = *(const v16bf*)(const void*)(Bp + bbase + 512);
        abase += 512;            // 32 lanes * 16
        bbase += bstep;
        acc0 = __builtin_amdgcn_wmma_f32_16x16x32_bf16(false, a, false, b0, (short)0, acc0, false, false);
        acc1 = __builtin_amdgcn_wmma_f32_16x16x32_bf16(false, a, false, b1, (short)0, acc1, false, false);
    }

    // epilogue (C/D layout: VGPR r -> M = r + 8*(lane>>4), N = lane&15)
    const int mhalf = lane >> 4;
    const int mlane = lane & 15;
    const int col0  = ntile * 32 + mlane;
    const int col1  = col0 + 16;
    #pragma unroll
    for (int r = 0; r < 8; ++r) {
        const int row = strip * 16 + r + 8 * mhalf;
        const float g = (gatev != nullptr) ? (scale * gatev[row]) : 0.f;
        float vals[2] = { acc0[r], acc1[r] };
        int   cols[2] = { col0, col1 };
        #pragma unroll
        for (int s = 0; s < 2; ++s) {
            const int c = cols[s];
            if (c >= ncols) continue;
            float val = vals[s] + (bias ? bias[c] : 0.f);
            if (act == 1) val = fmaxf(val, 0.f);
            if (resid) val = resid[(long)row * ldc + c] + val * g;
            if (outF) outF[(long)row * ldc + c] = val;
            if (outP) outP[packA_idx(row, c, Nout)] = f2bf(val);
            if (outR) outR[(long)row * Nout + c] = f2bf(val);
        }
    }
}

// ---------------------------------------------------------------- GRU elementwise fuse
// Writes h (f32, d_out), comm packed-A (Kd=256) for q/k/v GEMMs, and the
// head (cols 0..255) of gate_in packed-A (Kd=288).
__global__ void gru_kernel(const bf16_t* __restrict__ gib, const bf16_t* __restrict__ ghb,
                           const float* __restrict__ hprev, float* __restrict__ hout,
                           bf16_t* __restrict__ commP, bf16_t* __restrict__ gateinP)
{
    size_t idx = (size_t)blockIdx.x * 256 + threadIdx.x;
    if (idx >= (size_t)ROWS * HDIM) return;
    int row = (int)(idx >> 8); int u = (int)(idx & 255);
    size_t base = (size_t)row * 768;
    float ir = bf2f(gib[base + u]), iz = bf2f(gib[base + 256 + u]), in_ = bf2f(gib[base + 512 + u]);
    float hr = bf2f(ghb[base + u]), hz = bf2f(ghb[base + 256 + u]), hn  = bf2f(ghb[base + 512 + u]);
    float r = 1.f / (1.f + __expf(-(ir + hr)));
    float z = 1.f / (1.f + __expf(-(iz + hz)));
    float n = tanhf(in_ + r * hn);
    float h = (1.f - z) * n + z * hprev[idx];
    hout[idx] = h;
    bf16_t hb = f2bf(h);
    commP[packA_idx(row, u, HDIM)]   = hb;
    gateinP[packA_idx(row, u, GDIM)] = hb;
}

// ---------------------------------------------------------------- per-batch attention
__global__ void attention_kernel(const float* __restrict__ q, const float* __restrict__ k,
                                 const float* __restrict__ v, float* __restrict__ msgs)
{
    __shared__ float qs[HEADS][NA][HD];
    __shared__ float ks[HEADS][NA][HD];
    __shared__ float vs[HEADS][NA][VD];
    __shared__ float sc[HEADS][NA][NA];
    const int b = blockIdx.x;
    const int tid = threadIdx.x;            // 128 threads = 4 waves (one per head)
    for (int t = tid; t < NA * HEADS * HD; t += 128) {
        int j = t >> 6, c = t & 63;
        qs[c >> 4][j][c & 15] = q[(size_t)(b * NA + j) * 64 + c];
        ks[c >> 4][j][c & 15] = k[(size_t)(b * NA + j) * 64 + c];
    }
    for (int t = tid; t < NA * HEADS * VD; t += 128) {
        int j = t >> 5, c = t & 31;
        vs[c >> 3][j][c & 7] = v[(size_t)(b * NA + j) * 32 + c];
    }
    __syncthreads();

    const int h = tid >> 5;                  // head for this wave
    const int i = tid & 31;                  // agent row per lane
    for (int j = 0; j < NA; ++j) {
        float d = 0.f;
        #pragma unroll
        for (int e = 0; e < HD; ++e) d += qs[h][i][e] * ks[h][j][e];
        sc[h][i][j] = (j == i) ? -1e10f : d * 0.25f;   // 1/sqrt(16)
    }
    // top-8 selection (first pick is the max)
    unsigned mask = 0; float mmax = -1e30f;
    for (int t = 0; t < KTOP; ++t) {
        float best = -1e30f; int bj = 0;
        for (int j = 0; j < NA; ++j)
            if (!((mask >> j) & 1u)) { float s = sc[h][i][j]; if (s > best) { best = s; bj = j; } }
        mask |= 1u << bj;
        if (t == 0) mmax = best;
    }
    float sum = 0.f;
    for (int j = 0; j < NA; ++j)
        if ((mask >> j) & 1u) sum += __expf(sc[h][i][j] - mmax);
    const float inv = 1.f / sum;
    for (int j = 0; j < NA; ++j)
        sc[h][i][j] = ((mask >> j) & 1u) ? __expf(sc[h][i][j] - mmax) * inv : 0.f;

    float acc[VD];
    #pragma unroll
    for (int d = 0; d < VD; ++d) acc[d] = 0.f;
    for (int j = 0; j < NA; ++j) {
        float w = sc[h][i][j];
        #pragma unroll
        for (int d = 0; d < VD; ++d) acc[d] += w * vs[h][j][d];
    }
    #pragma unroll
    for (int d = 0; d < VD; ++d)
        msgs[(size_t)(b * NA + i) * FC + h * VD + d] = acc[d];
}

// ---------------------------------------------------------------- layernorm over FC
// Writes normed packed-A (Kd=32) and tail (cols 256..287) of gate_in packed-A (Kd=288).
__global__ void ln_kernel(const float* __restrict__ msgs, const float* __restrict__ lng,
                          const float* __restrict__ lnb, bf16_t* __restrict__ normedP,
                          bf16_t* __restrict__ gateinP)
{
    int row = (int)((size_t)blockIdx.x * 256 + threadIdx.x);
    if (row >= ROWS) return;
    float x[FC]; float mu = 0.f;
    #pragma unroll
    for (int c = 0; c < FC; ++c) { x[c] = msgs[(size_t)row * FC + c]; mu += x[c]; }
    mu *= (1.f / FC);
    float var = 0.f;
    #pragma unroll
    for (int c = 0; c < FC; ++c) { float d = x[c] - mu; var += d * d; }
    var *= (1.f / FC);
    float rs = rsqrtf(var + 1e-5f);
    #pragma unroll
    for (int c = 0; c < FC; ++c) {
        float nv = (x[c] - mu) * rs * lng[c] + lnb[c];
        bf16_t bv = f2bf(nv);
        normedP[packA_idx(row, c, FC)]         = bv;
        gateinP[packA_idx(row, 256 + c, GDIM)] = bv;
    }
}

// ---------------------------------------------------------------- gate = sigmoid(g1 @ Wg2 + bg2)
// wave-per-row: 8 bf16 per lane + shuffle reduction
__global__ void gate_kernel(const bf16_t* __restrict__ g1b, const float* __restrict__ Wg2,
                            const float* __restrict__ bg2, float* __restrict__ gate)
{
    const int row  = (int)(((size_t)blockIdx.x * 256 + threadIdx.x) >> 5);
    const int lane = threadIdx.x & 31;
    if (row >= ROWS) return;
    const bf16_t* gp = g1b + (size_t)row * HDIM + lane * 8;
    const float*  wp = Wg2 + lane * 8;
    float s = 0.f;
    #pragma unroll
    for (int e = 0; e < 8; ++e) s += bf2f(gp[e]) * wp[e];
    #pragma unroll
    for (int off = 16; off > 0; off >>= 1) s += __shfl_down(s, off, 32);
    if (lane == 0) gate[row] = 1.f / (1.f + __expf(-(s + bg2[0])));
}

// ================================================================ host
extern "C" void kernel_launch(void* const* d_in, const int* in_sizes, int n_in,
                              void* d_out, int out_size, void* d_ws, size_t ws_size,
                              hipStream_t stream)
{
    (void)in_sizes; (void)n_in; (void)out_size; (void)ws_size;
    const float* inputs = (const float*)d_in[0];
    const float* hidden = (const float*)d_in[1];
    const float* W1  = (const float*)d_in[2];   const float* b1  = (const float*)d_in[3];
    const float* Wih = (const float*)d_in[4];   const float* bih = (const float*)d_in[5];
    const float* Whh = (const float*)d_in[6];   const float* bhh = (const float*)d_in[7];
    const float* Wq  = (const float*)d_in[8];   const float* bq  = (const float*)d_in[9];
    const float* Wk  = (const float*)d_in[10];  const float* bk  = (const float*)d_in[11];
    const float* Wv  = (const float*)d_in[12];  const float* bv  = (const float*)d_in[13];
    const float* lng = (const float*)d_in[14];  const float* lnb = (const float*)d_in[15];
    const float* Wg1 = (const float*)d_in[16];  const float* bg1 = (const float*)d_in[17];
    const float* Wg2 = (const float*)d_in[18];  const float* bg2 = (const float*)d_in[19];
    const float* Wr  = (const float*)d_in[20];  const float* br  = (const float*)d_in[21];
    const float* Wp1 = (const float*)d_in[22];  const float* bp1 = (const float*)d_in[23];
    const float* Wp2 = (const float*)d_in[24];  const float* bp2 = (const float*)d_in[25];

    float* logits = (float*)d_out;                           // [ROWS, 20]
    float* hout   = (float*)d_out + (size_t)ROWS * ACTD;     // [ROWS, 256]

    // ---- workspace carve (256B aligned regions)
    char* p = (char*)d_ws;
    auto carve = [&](size_t bytes) -> char* {
        char* r = p; p += (bytes + 255) & ~(size_t)255; return r;
    };
    bf16_t* inP     = (bf16_t*)carve((size_t)ROWS * DIN * 2);    // packed-A Kd=128
    bf16_t* hP      = (bf16_t*)carve((size_t)ROWS * HDIM * 2);   // packed-A Kd=256
    bf16_t* xP      = (bf16_t*)carve((size_t)ROWS * HDIM * 2);   // packed-A Kd=256
    bf16_t* gib     = (bf16_t*)carve((size_t)ROWS * 768 * 2);    // row-major
    bf16_t* ghb     = (bf16_t*)carve((size_t)ROWS * 768 * 2);    // row-major
    bf16_t* commP   = (bf16_t*)carve((size_t)ROWS * HDIM * 2);   // packed-A Kd=256
    float*  qf      = (float*) carve((size_t)ROWS * 64 * 4);
    float*  kf      = (float*) carve((size_t)ROWS * 64 * 4);
    float*  vf      = (float*) carve((size_t)ROWS * 32 * 4);
    float*  msgs    = (float*) carve((size_t)ROWS * FC * 4);
    bf16_t* normedP = (bf16_t*)carve((size_t)ROWS * FC * 2);     // packed-A Kd=32
    bf16_t* gateinP = (bf16_t*)carve((size_t)ROWS * GDIM * 2);   // packed-A Kd=288
    bf16_t* g1b     = (bf16_t*)carve((size_t)ROWS * HDIM * 2);   // row-major
    float*  gatev   = (float*) carve((size_t)ROWS * 4);
    bf16_t* fusedP  = (bf16_t*)carve((size_t)ROWS * HDIM * 2);   // packed-A Kd=256
    bf16_t* p1P     = (bf16_t*)carve((size_t)ROWS * HDIM * 2);   // packed-A Kd=256
    bf16_t* w1b     = (bf16_t*)carve((size_t)DIN * HDIM * 2);
    bf16_t* wihb    = (bf16_t*)carve((size_t)HDIM * 768 * 2);
    bf16_t* whhb    = (bf16_t*)carve((size_t)HDIM * 768 * 2);
    bf16_t* wqb     = (bf16_t*)carve((size_t)HDIM * 64 * 2);
    bf16_t* wkb     = (bf16_t*)carve((size_t)HDIM * 64 * 2);
    bf16_t* wvb     = (bf16_t*)carve((size_t)HDIM * 32 * 2);
    bf16_t* wg1b    = (bf16_t*)carve((size_t)GDIM * HDIM * 2);
    bf16_t* wrb     = (bf16_t*)carve((size_t)FC * HDIM * 2);
    bf16_t* wp1b    = (bf16_t*)carve((size_t)HDIM * HDIM * 2);
    bf16_t* wp2b    = (bf16_t*)carve((size_t)HDIM * 32 * 2);

    auto cvtA = [&](const float* s, bf16_t* d, int M, int Kd) {
        long n = (long)M * Kd;
        cvt_packA<<<(int)((n + 255) / 256), 256, 0, stream>>>(s, d, M, Kd);
    };
    auto packW = [&](const float* w, bf16_t* d, int Kd, int Nout, int srcCols) {
        long n = (long)Kd * Nout;
        pack_w<<<(int)((n + 255) / 256), 256, 0, stream>>>(w, d, Kd, Nout, srcCols);
    };
    auto gemm = [&](const bf16_t* A, const bf16_t* B, const float* bias,
                    int Kd, int Nout, float* oF, int ldc, int ncols,
                    bf16_t* oP, bf16_t* oR, int act,
                    const float* resid, const float* gv, float scale) {
        long tiles = (long)(ROWS / 16) * (Nout / 32);
        int blocks = (int)((tiles + 7) / 8);
        gemm_bf16_wmma<<<blocks, 256, 0, stream>>>(A, B, bias, ROWS, Kd, Nout,
                                                   oF, ldc, ncols, oP, oR, act, resid, gv, scale);
    };

    // ---- convert activations + weights into WMMA fragment layouts
    cvtA(inputs, inP, ROWS, DIN);
    cvtA(hidden, hP,  ROWS, HDIM);
    packW(W1,  w1b,  DIN,  HDIM, HDIM);
    packW(Wih, wihb, HDIM, 768,  768);
    packW(Whh, whhb, HDIM, 768,  768);
    packW(Wq,  wqb,  HDIM, 64,   64);
    packW(Wk,  wkb,  HDIM, 64,   64);
    packW(Wv,  wvb,  HDIM, 32,   32);
    packW(Wg1, wg1b, GDIM, HDIM, HDIM);
    packW(Wr,  wrb,  FC,   HDIM, HDIM);
    packW(Wp1, wp1b, HDIM, HDIM, HDIM);
    packW(Wp2, wp2b, HDIM, 32,   ACTD);   // zero-padded 20 -> 32

    // ---- backbone: x = relu(inputs @ W1 + b1) -> packed
    gemm(inP, w1b, b1, DIN, HDIM, nullptr, 0, HDIM, xP, nullptr, 1, nullptr, nullptr, 0.f);
    // ---- GRU gate GEMMs -> row-major bf16
    gemm(xP, wihb, bih, HDIM, 768, nullptr, 0, 768, nullptr, gib, 0, nullptr, nullptr, 0.f);
    gemm(hP, whhb, bhh, HDIM, 768, nullptr, 0, 768, nullptr, ghb, 0, nullptr, nullptr, 0.f);
    // ---- GRU elementwise -> h (f32, d_out), comm packed, gate_in head
    {
        long n = (long)ROWS * HDIM;
        gru_kernel<<<(int)((n + 255) / 256), 256, 0, stream>>>(gib, ghb, hidden, hout, commP, gateinP);
    }
    // ---- q/k/v projections -> f32 row-major
    gemm(commP, wqb, bq, HDIM, 64, qf, 64, 64, nullptr, nullptr, 0, nullptr, nullptr, 0.f);
    gemm(commP, wkb, bk, HDIM, 64, kf, 64, 64, nullptr, nullptr, 0, nullptr, nullptr, 0.f);
    gemm(commP, wvb, bv, HDIM, 32, vf, 32, 32, nullptr, nullptr, 0, nullptr, nullptr, 0.f);
    // ---- attention (per batch block)
    attention_kernel<<<BS_, 128, 0, stream>>>(qf, kf, vf, msgs);
    // ---- layernorm -> normed packed + gate_in tail
    ln_kernel<<<(ROWS + 255) / 256, 256, 0, stream>>>(msgs, lng, lnb, normedP, gateinP);
    // ---- gate MLP
    gemm(gateinP, wg1b, bg1, GDIM, HDIM, nullptr, 0, HDIM, nullptr, g1b, 1, nullptr, nullptr, 0.f);
    gate_kernel<<<(int)(((size_t)ROWS * 32 + 255) / 256), 256, 0, stream>>>(g1b, Wg2, bg2, gatev);
    // ---- residual adapter fused with gate: fused = h + (normed @ Wr + br) * (0.1 * gate)
    gemm(normedP, wrb, br, FC, HDIM, nullptr, HDIM, HDIM, fusedP, nullptr, 0, hout, gatev, 0.1f);
    // ---- policy head
    gemm(fusedP, wp1b, bp1, HDIM, HDIM, nullptr, 0, HDIM, p1P, nullptr, 1, nullptr, nullptr, 0.f);
    gemm(p1P, wp2b, bp2, HDIM, 32, logits, ACTD, ACTD, nullptr, nullptr, 0, nullptr, nullptr, 0.f);
}
